// EmotionAttentionLayer_14070312861867
// MI455X (gfx1250) — matrix operand order repaired and checked by reading, not compiled, not run on hardware
//
#include <hip/hip_runtime.h>
#include <hip/hip_bf16.h>

typedef _Float16 half8   __attribute__((ext_vector_type(8)));
typedef _Float16 half16  __attribute__((ext_vector_type(16)));
typedef float    floatx8 __attribute__((ext_vector_type(8)));

#define B_      4
#define T_      2048
#define U_      512
#define H_      8
#define DH_     64
#define MTOT_   (B_ * T_)   // 8192 rows for the projections

// ---- WMMA fragment loaders per CDNA5 ISA 7.12.2 (wave32) ------------------
// A (16x32 f16): lane row = lane&15; halves j0..7 / j8..15 map to k chunks
//   {0..7,16..23} (lanes 0-15) or {8..15,24..31} (lanes 16-31).
// caller passes p = rowBase + ((lane>>4)<<3)
static __device__ inline half16 load_a_frag(const _Float16* p) {
    half8 lo = *(const half8*)(p);
    half8 hi = *(const half8*)(p + 16);
    return __builtin_shufflevector(lo, hi, 0,1,2,3,4,5,6,7,8,9,10,11,12,13,14,15);
}
// B (32x16 f16): col n = lane&15; k chunks {0..7,16..23} (lanes 0-15) or {8..15,24..31} (lanes 16-31)
// caller passes p = colBase + ((lane>>4)<<3)
static __device__ inline half16 load_b_frag(const _Float16* p) {
    half8 lo = *(const half8*)(p);
    half8 hi = *(const half8*)(p + 16);
    return __builtin_shufflevector(lo, hi, 0,1,2,3,4,5,6,7,8,9,10,11,12,13,14,15);
}

static __device__ inline floatx8 wmma_f16(half16 a, half16 b, floatx8 c) {
    return __builtin_amdgcn_wmma_f32_16x16x32_f16(false, a, false, b, (short)0, c,
                                                  false, false);
}

// ---- CDNA5 async-to-LDS path (ASYNCcnt-tracked DMA into LDS) --------------
static __device__ inline unsigned lds_off_u32(const void* p) {
    return (unsigned)(unsigned long long)
           (const __attribute__((address_space(3))) char*)p;
}
static __device__ inline void async_copy_b128(unsigned lds_byte, const void* g) {
    asm volatile("global_load_async_to_lds_b128 %0, %1, off"
                 :: "v"(lds_byte), "v"((unsigned long long)(size_t)g)
                 : "memory");
}
static __device__ inline void wait_async0() {
    asm volatile("s_wait_asynccnt 0x0" ::: "memory");
}

// ---------------------------------------------------------------------------
// Kernel 0: streaming fp32 -> fp16 conversion (8 elements/thread, b128 in/out)
// ---------------------------------------------------------------------------
__global__ __launch_bounds__(256) void cvt_f16_kernel(
    const float* __restrict__ in, _Float16* __restrict__ out, int n)
{
    const int i = (blockIdx.x * 256 + threadIdx.x) * 8;
    if (i + 8 > n) return;
    const float4 a = ((const float4*)(in + i))[0];
    const float4 b = ((const float4*)(in + i))[1];
    half8 h;
    h[0] = (_Float16)a.x; h[1] = (_Float16)a.y;
    h[2] = (_Float16)a.z; h[3] = (_Float16)a.w;
    h[4] = (_Float16)b.x; h[5] = (_Float16)b.y;
    h[6] = (_Float16)b.z; h[7] = (_Float16)b.w;
    *(half8*)(out + i) = h;
}

// ---------------------------------------------------------------------------
// Kernel 1: Y = relu(Xh @ Wh^T + bias) -> fp16, all-fp16 inputs.
// 256 thr (8 waves, 4(M) x 2(N)), block tile 128x128, wave tile 32x64.
// Async-to-LDS ping-pong tiles, ONE barrier per k-step, no VGPR staging.
// ---------------------------------------------------------------------------
#define AS_STRIDE 40   // 32 + 8 pad halves; 80B rows (16B aligned)
__global__ __launch_bounds__(256) void proj_relu_kernel(
    const _Float16* __restrict__ Xh, const _Float16* __restrict__ Wh,
    const float* __restrict__ bias, _Float16* __restrict__ Y)
{
    __shared__ __attribute__((aligned(16))) _Float16 As[2][128 * AS_STRIDE];
    __shared__ __attribute__((aligned(16))) _Float16 Bs[2][128 * AS_STRIDE];

    const int tid  = threadIdx.x;
    const int lane = tid & 31;
    const int wv   = tid >> 5;
    const int wm   = wv >> 1;     // 0..3
    const int wn   = wv & 1;      // 0..1
    const int blockM = blockIdx.x * 128;
    const int blockN = blockIdx.y * 128;

    floatx8 acc[2][4];
#pragma unroll
    for (int i = 0; i < 2; ++i)
#pragma unroll
        for (int j = 0; j < 4; ++j) acc[i][j] = (floatx8)0.0f;

    // DMA one 128x32-half tile of X and W each: 512 chunks x 16B per matrix
    auto tile_async = [&](int k0, int buf) {
        const unsigned ab = lds_off_u32(&As[buf][0]);
        const unsigned bb = lds_off_u32(&Bs[buf][0]);
#pragma unroll
        for (int ii = 0; ii < 2; ++ii) {
            const int c   = tid + ii * 256;
            const int row = c >> 2;
            const int ch  = c & 3;
            const unsigned loff = (unsigned)(row * (AS_STRIDE * 2) + ch * 16);
            async_copy_b128(ab + loff, Xh + (size_t)(blockM + row) * U_ + k0 + ch * 8);
            async_copy_b128(bb + loff, Wh + (size_t)(blockN + row) * U_ + k0 + ch * 8);
        }
    };

    constexpr int KSTEPS = U_ / 32;     // 16
    tile_async(0, 0);
    wait_async0();
    tile_async(32, 1);
    __syncthreads();

    for (int it = 0; it < KSTEPS; ++it) {
        const int cur = it & 1;
        // ---- compute from buffer `cur` ----
        half16 bf[4];
#pragma unroll
        for (int nf = 0; nf < 4; ++nf) {
            const _Float16* p = &Bs[cur][(wn * 64 + nf * 16 + (lane & 15)) * AS_STRIDE
                                         + ((lane >> 4) << 3)];
            bf[nf] = load_b_frag(p);
        }
#pragma unroll
        for (int mi = 0; mi < 2; ++mi) {
            const _Float16* p = &As[cur][(wm * 32 + mi * 16 + (lane & 15)) * AS_STRIDE
                                         + ((lane >> 4) << 3)];
            half16 af = load_a_frag(p);
#pragma unroll
            for (int nf = 0; nf < 4; ++nf)
                acc[mi][nf] = wmma_f16(af, bf[nf], acc[mi][nf]);
        }
        wait_async0();                  // tile it+1 DMA complete (issued last iter)
        __syncthreads();                // single barrier per k-step
        if (it + 2 < KSTEPS) tile_async((it + 2) * 32, cur);  // refill freed buffer
    }

    // epilogue: bias + relu, write fp16
    const int rowHalf = (lane >> 4) << 3;
#pragma unroll
    for (int mi = 0; mi < 2; ++mi) {
#pragma unroll
        for (int nf = 0; nf < 4; ++nf) {
            const int col = blockN + wn * 64 + nf * 16 + (lane & 15);
            const float bb = bias[col];
#pragma unroll
            for (int r = 0; r < 8; ++r) {
                const int row = blockM + wm * 32 + mi * 16 + r + rowHalf;
                float v = acc[mi][nf][r] + bb;
                v = v > 0.0f ? v : 0.0f;
                Y[(size_t)row * U_ + col] = (_Float16)v;
            }
        }
    }
}

// ---------------------------------------------------------------------------
// Kernel 2: flash attention. grid = (T/128, B*H), block = 256 (8 waves).
// Wave owns 16 query rows; 64-key tiles; online softmax in fp32.
// K tiles: global_load_async_to_lds_b128 ping-pong (ASYNCcnt).
// V tiles: register-staged transposed store. ONE barrier per tile.
// ---------------------------------------------------------------------------
#define KV_STRIDE 72   // 64 + 8 pad halves; 144B rows (16B aligned)
__global__ __launch_bounds__(256) void flash_attn_kernel(
    const _Float16* __restrict__ Qh, const _Float16* __restrict__ Kh,
    const _Float16* __restrict__ Vh, const float* __restrict__ queries,
    float* __restrict__ out)
{
    __shared__ __attribute__((aligned(16))) _Float16 Ks[2][64 * KV_STRIDE]; // [key][d]
    __shared__ __attribute__((aligned(16))) _Float16 Vs[2][64 * KV_STRIDE]; // [d][key]
    __shared__ __attribute__((aligned(16))) _Float16 Ps[8 * 16 * KV_STRIDE];

    const int tid  = threadIdx.x;
    const int lane = tid & 31;
    const int wv   = tid >> 5;
    const int b    = blockIdx.y >> 3;   // / H
    const int h    = blockIdx.y & 7;
    const int qb   = blockIdx.x * 128 + wv * 16;
    const size_t baseBT = (size_t)b * T_;

    // Q fragments (16 rows x 64 dh), pre-scaled by 1/sqrt(Dh) = 0.125
    half16 qa[2];
    {
        const int t = qb + (lane & 15);
        const _Float16* p = Qh + (baseBT + t) * U_ + h * DH_ + ((lane >> 4) << 3);
        qa[0] = load_a_frag(p);
        qa[1] = load_a_frag(p + 32);
#pragma unroll
        for (int j = 0; j < 16; ++j) {
            qa[0][j] *= (_Float16)0.125f;
            qa[1][j] *= (_Float16)0.125f;
        }
    }

    floatx8 o[4];
#pragma unroll
    for (int nf = 0; nf < 4; ++nf) o[nf] = (floatx8)0.0f;
    float m_r[8], l_r[8];
#pragma unroll
    for (int r = 0; r < 8; ++r) { m_r[r] = -1e30f; l_r[r] = 0.0f; }

    const int lrow = tid >> 2;          // 0..63 (key)
    const int lcol = (tid & 3) * 16;    // 0/16/32/48 (d)

    half8 va, vb;                       // V staging registers (next tile)
    auto vload = [&](int kt) {
        const _Float16* s = Vh + (baseBT + kt + lrow) * U_ + h * DH_ + lcol;
        va = *(const half8*)(s);
        vb = *(const half8*)(s + 8);
    };
    auto vstore = [&](int buf) {        // transposed: Vs[d][key]
#pragma unroll
        for (int j = 0; j < 8; ++j) Vs[buf][(lcol + j) * KV_STRIDE + lrow] = va[j];
#pragma unroll
        for (int j = 0; j < 8; ++j) Vs[buf][(lcol + 8 + j) * KV_STRIDE + lrow] = vb[j];
    };
    auto kasync = [&](int kt, int buf) { // K tile: 512 x 16B chunks, 2 per thread
        const unsigned base = lds_off_u32(&Ks[buf][0]);
#pragma unroll
        for (int ii = 0; ii < 2; ++ii) {
            const int c   = tid + ii * 256;
            const int row = c >> 3;
            const int ch  = c & 7;
            const _Float16* g = Kh + (baseBT + kt + row) * U_ + h * DH_ + ch * 8;
            async_copy_b128(base + (unsigned)(row * (KV_STRIDE * 2) + ch * 16), g);
        }
    };

    constexpr int NT = T_ / 64;         // 32 key tiles

    // ---- pipeline preamble ----
    kasync(0, 0);
    vload(0);
    vstore(0);
    vload(64);
    wait_async0();
    kasync(64, 1);
    __syncthreads();

    for (int it = 0; it < NT; ++it) {
        const int cur = it & 1;
        if (it + 1 < NT) {
            vstore(cur ^ 1);                       // tile it+1 into free buffer
            if (it + 2 < NT) vload((it + 2) * 64); // issue next V loads
        }

        // ---- S = Q K^T (16 x 64), scale folded into qa ----
        floatx8 s[4];
#pragma unroll
        for (int nf = 0; nf < 4; ++nf) {
            const _Float16* p = &Ks[cur][(nf * 16 + (lane & 15)) * KV_STRIDE
                                         + ((lane >> 4) << 3)];
            s[nf] = (floatx8)0.0f;
            s[nf] = wmma_f16(qa[0], load_b_frag(p),      s[nf]);
            s[nf] = wmma_f16(qa[1], load_b_frag(p + 32), s[nf]);
        }

        // ---- online softmax (row = r + 8*(lane>=16); cols across 16-lane half)
#pragma unroll
        for (int r = 0; r < 8; ++r) {
            float mx = fmaxf(fmaxf(s[0][r], s[1][r]), fmaxf(s[2][r], s[3][r]));
#pragma unroll
            for (int off = 8; off >= 1; off >>= 1)
                mx = fmaxf(mx, __shfl_xor(mx, off, 32));
            const float mnew  = fmaxf(m_r[r], mx);
            const float alpha = __expf(m_r[r] - mnew);
            m_r[r] = mnew;
            float sum = 0.0f;
#pragma unroll
            for (int nf = 0; nf < 4; ++nf) {
                const float e = __expf(s[nf][r] - mnew);
                s[nf][r] = e;
                sum += e;
            }
#pragma unroll
            for (int off = 8; off >= 1; off >>= 1)
                sum += __shfl_xor(sum, off, 32);
            l_r[r] = l_r[r] * alpha + sum;
#pragma unroll
            for (int nf = 0; nf < 4; ++nf) o[nf][r] *= alpha;
        }

        // ---- C-layout -> A-layout for P via per-wave LDS staging ----
        _Float16* pw = Ps + wv * (16 * KV_STRIDE);
        const int prow = (lane >> 4) << 3;
#pragma unroll
        for (int r = 0; r < 8; ++r)
#pragma unroll
            for (int nf = 0; nf < 4; ++nf)
                pw[(prow + r) * KV_STRIDE + nf * 16 + (lane & 15)] = (_Float16)s[nf][r];

        const _Float16* pA = pw + (lane & 15) * KV_STRIDE + ((lane >> 4) << 3);
        half16 pa0 = load_a_frag(pA);
        half16 pa1 = load_a_frag(pA + 32);

        // ---- O += P V ----
#pragma unroll
        for (int nf = 0; nf < 4; ++nf) {
            const _Float16* vp = &Vs[cur][(nf * 16 + (lane & 15)) * KV_STRIDE
                                          + ((lane >> 4) << 3)];
            o[nf] = wmma_f16(pa0, load_b_frag(vp),      o[nf]);
            o[nf] = wmma_f16(pa1, load_b_frag(vp + 32), o[nf]);
        }

        wait_async0();                  // own K-async (tile it+1) complete
        __syncthreads();                // single barrier per tile
        if (it + 2 < NT) kasync((it + 2) * 64, cur);  // refill just-freed buffer
    }

    // ---- epilogue: normalize, add residual, fp32 store ----
#pragma unroll
    for (int r = 0; r < 8; ++r) {
        const float inv = 1.0f / l_r[r];
        const int t = qb + r + ((lane >> 4) << 3);
#pragma unroll
        for (int nf = 0; nf < 4; ++nf) {
            const size_t idx = (baseBT + t) * U_ + h * DH_ + nf * 16 + (lane & 15);
            out[idx] = o[nf][r] * inv + queries[idx];
        }
    }
}

// ---------------------------------------------------------------------------
extern "C" void kernel_launch(void* const* d_in, const int* in_sizes, int n_in,
                              void* d_out, int out_size, void* d_ws, size_t ws_size,
                              hipStream_t stream) {
    const float* queries = (const float*)d_in[0];
    const float* keys    = (const float*)d_in[1];
    const float* values  = (const float*)d_in[2];
    const float* Wq      = (const float*)d_in[3];
    const float* bq      = (const float*)d_in[4];
    const float* Wk      = (const float*)d_in[5];
    const float* bk      = (const float*)d_in[6];
    const float* Wv      = (const float*)d_in[7];
    const float* bv      = (const float*)d_in[8];
    float* out = (float*)d_out;

    const size_t NU = (size_t)MTOT_ * U_;   // 4,194,304
    const size_t WW = (size_t)U_ * U_;      //   262,144
    _Float16* Qh  = (_Float16*)d_ws;        // projection outputs (fp16)
    _Float16* Kh  = Qh + NU;
    _Float16* Vh  = Kh + NU;
    _Float16* Xq  = Vh + NU;                // converted inputs (fp16)
    _Float16* Xk  = Xq + NU;
    _Float16* Xv  = Xk + NU;
    _Float16* Wqh = Xv + NU;                // converted weights (fp16)
    _Float16* Wkh = Wqh + WW;
    _Float16* Wvh = Wkh + WW;

    dim3 blk(256);
    // fp32 -> fp16 passes (8 elems/thread)
    cvt_f16_kernel<<<dim3(NU / 2048), blk, 0, stream>>>(queries, Xq, (int)NU);
    cvt_f16_kernel<<<dim3(NU / 2048), blk, 0, stream>>>(keys,    Xk, (int)NU);
    cvt_f16_kernel<<<dim3(NU / 2048), blk, 0, stream>>>(values,  Xv, (int)NU);
    cvt_f16_kernel<<<dim3(WW / 2048), blk, 0, stream>>>(Wq, Wqh, (int)WW);
    cvt_f16_kernel<<<dim3(WW / 2048), blk, 0, stream>>>(Wk, Wkh, (int)WW);
    cvt_f16_kernel<<<dim3(WW / 2048), blk, 0, stream>>>(Wv, Wvh, (int)WW);

    dim3 gGemm(MTOT_ / 128, U_ / 128);
    proj_relu_kernel<<<gGemm, blk, 0, stream>>>(Xq, Wqh, bq, Qh);
    proj_relu_kernel<<<gGemm, blk, 0, stream>>>(Xk, Wkh, bk, Kh);
    proj_relu_kernel<<<gGemm, blk, 0, stream>>>(Xv, Wvh, bv, Vh);

    dim3 gAttn(T_ / 128, B_ * H_);
    flash_attn_kernel<<<gAttn, blk, 0, stream>>>(Qh, Kh, Vh, queries, out);
}
